// LSTMNet_62964220559617
// MI455X (gfx1250) — matrix operand order repaired and checked by reading, not compiled, not run on hardware
//
#include <hip/hip_runtime.h>
#include <hip/hip_bf16.h>
#include <stdint.h>

// Problem sizes (fixed by the reference).
#define BB 128
#define TT 512
#define VV 256
#define HH 2048
#define G4 (4 * HH)   // 8192 gate columns

typedef __attribute__((ext_vector_type(16))) __bf16 v16bf;
typedef __attribute__((ext_vector_type(8)))  __bf16 v8bf;
typedef __attribute__((ext_vector_type(8)))  float  v8f;

union V16U { v16bf v; v8bf h[2]; };

// ---------------------------------------------------------------------------
// CDNA5 async global->LDS copy helpers (ASYNCcnt-tracked DMA, ISA ch.10/15.18)
// ---------------------------------------------------------------------------
// Low 32 bits of a generic pointer to __shared__ are the LDS byte address
// (flat aperture mapping).
__device__ __forceinline__ uint32_t lds_off(const void* p) {
  return (uint32_t)(uintptr_t)p;
}

// 16B per-lane async copy: LDS[vdst] = MEM[vaddr], tracked with ASYNCcnt.
__device__ __forceinline__ void async_b128(uint32_t lds, const __bf16* g) {
  asm volatile("global_load_async_to_lds_b128 %0, %1, off"
               :: "v"(lds), "v"(g) : "memory");
}

template <int N>
__device__ __forceinline__ void wait_async() {
  asm volatile("s_wait_asynccnt %0" :: "i"(N) : "memory");
}

// ---------------------------------------------------------------------------
// WMMA fragment loaders (bf16, 16x16x32, wave32)
// ---------------------------------------------------------------------------
// A-matrix 16x32 bf16 layout (ISA 7.12.2):
//   lane m (0-15):  V0-3 = K[k0..k0+7],    V4-7 = K[k0+16..k0+23]
//   lane m+16:      V0-3 = K[k0+8..k0+15], V4-7 = K[k0+24..k0+31]
__device__ __forceinline__ v16bf wmma_load_a(const __bf16* base, int lda, int k0, int lane) {
  const int m = lane & 15;
  const int q = (lane >> 4) * 8;
  const __bf16* p = base + (size_t)m * lda + k0 + q;
  V16U u;
  u.h[0] = *(const v8bf*)(p);
  u.h[1] = *(const v8bf*)(p + 16);
  return u.v;
}

// B-matrix 32x16 bf16: lane n (0-15) holds K[k0..k0+7], K[k0+16..k0+23] of column n,
// lane n+16 holds K[k0+8..k0+15], K[k0+24..k0+31].  Column n of B == row (row0+n) of W
// (we compute X @ W^T, W row-major -> two 16-byte runs per lane).
__device__ __forceinline__ v16bf wmma_load_b(const __bf16* W, int row0, int ldw, int k0, int lane) {
  const int n  = lane & 15;
  const int kq = (lane >> 4) * 8;
  const __bf16* p = W + (size_t)(row0 + n) * ldw + k0 + kq;
  V16U u;
  u.h[0] = *(const v8bf*)(p);
  u.h[1] = *(const v8bf*)(p + 16);
  return u.v;
}

__device__ __forceinline__ v8f wmma_bf16(v16bf a, v16bf b, v8f c) {
  return __builtin_amdgcn_wmma_f32_16x16x32_bf16(false, a, false, b, (short)0, c, false, false);
}

__device__ __forceinline__ float sigm(float v)   { return 1.0f / (1.0f + __expf(-v)); }
__device__ __forceinline__ float tanh_f(float v) { return 2.0f * sigm(2.0f * v) - 1.0f; }

// ---------------------------------------------------------------------------
// Prep kernels
// ---------------------------------------------------------------------------
__global__ void k_f32_to_bf16(const float* __restrict__ src, __bf16* __restrict__ dst, int n) {
  int i = blockIdx.x * blockDim.x + threadIdx.x;
  if (i < n) dst[i] = (__bf16)src[i];
}

__global__ void k_bias_sum(const float* __restrict__ a, const float* __restrict__ b,
                           float* __restrict__ o, int n) {
  int i = blockIdx.x * blockDim.x + threadIdx.x;
  if (i < n) o[i] = a[i] + b[i];
}

__global__ void k_zero_state(__bf16* __restrict__ h0, float* __restrict__ c, int n) {
  int i = blockIdx.x * blockDim.x + threadIdx.x;
  if (i < n) { h0[i] = (__bf16)0.0f; c[i] = 0.0f; }
}

// ---------------------------------------------------------------------------
// One LSTM timestep.  Block owns a 32-row x 64-col tile of ALL four gates
// (i/f/g/o fusion is block-local).  K = 2048 (h @ Whh^T) + 256 (x_t @ Wih^T),
// consumed in 18 double-buffered async-staged chunks of K=128.
// Pipeline per stage (ONE barrier): wait(own copy) -> barrier -> issue(next)
// -> compute(current).  issue(s+1) overlaps compute(s); buffer (s+1)&1 was
// last read in compute(s-1), which every thread finished before this barrier.
//   grid  = (HH/64, BB/32) = (32, 4)
//   block = 256 threads = 8 waves;  wave w: gate (w&3), column half (w>>2)
// ---------------------------------------------------------------------------
#define STEP_KSTAGE 128
#define STEP_LDA    (STEP_KSTAGE + 8)   // pad keeps 16B alignment, skews banks

__global__ __launch_bounds__(256) void lstm_step(
    const __bf16* __restrict__ x_b,    // [B, T, V] bf16
    const __bf16* __restrict__ Wih,    // [4H, V]   bf16
    const __bf16* __restrict__ Whh,    // [4H, H]   bf16
    const float*  __restrict__ bias,   // [4H]      fp32 (b_ih + b_hh)
    const __bf16* __restrict__ h_in,   // [B, H]    state after step t-1
    float*        __restrict__ c,      // [B, H]    in/out, block-exclusive
    __bf16*       __restrict__ h_out,  // [B, H]    state after step t
    int t)
{
  __shared__ __bf16 tileA[2][32 * STEP_LDA];   // 2 x 8.5 KB staged A chunks
  __shared__ float  gates[4][32][65];          // ~32.5 KB gate tile (padded)

  const int tid  = threadIdx.x;
  const int wave = tid >> 5;
  const int lane = tid & 31;
  const int g    = wave & 3;                   // gate: 0=i 1=f 2=g 3=o
  const int s    = wave >> 2;                  // column half (0/1) of 64
  const int hid0 = blockIdx.x * 64;            // hidden column base
  const int m0   = blockIdx.y * 32;            // batch row base
  const int colW0 = g * HH + hid0 + s * 32;    // base row in Wih/Whh

  // Staging map: thread -> (row sm, 16-element segment sk); 2 x b128 each.
  const int sm = tid >> 3;                     // 0..31
  const int sk = (tid & 7) * 16;               // 0..112
  const __bf16* aRow_h = h_in + (size_t)(m0 + sm) * HH + sk;
  const __bf16* aRow_x = x_b + ((size_t)(m0 + sm) * TT + t) * VV + sk;
  const uint32_t ldsA[2] = { lds_off(&tileA[0][sm * STEP_LDA + sk]),
                             lds_off(&tileA[1][sm * STEP_LDA + sk]) };

  v8f acc[2][2] = {};                          // [m-tile][n-tile]

  const int NSTAGE_H = HH / STEP_KSTAGE;             // 16
  const int NSTAGE   = NSTAGE_H + VV / STEP_KSTAGE;  // 18

  auto issue = [&](int stg) {
    const __bf16* src = (stg < NSTAGE_H)
        ? (aRow_h + stg * STEP_KSTAGE)
        : (aRow_x + (stg - NSTAGE_H) * STEP_KSTAGE);
    const uint32_t l = ldsA[stg & 1];
    async_b128(l,      src);
    async_b128(l + 16, src + 8);
  };

  issue(0);
  for (int stg = 0; stg < NSTAGE; ++stg) {
    wait_async<0>();          // own slice of stage stg has landed in LDS
    __syncthreads();          // everyone's slice landed; prev compute done
    if (stg + 1 < NSTAGE) issue(stg + 1);   // overlaps compute(stg)

    const __bf16* At  = tileA[stg & 1];
    const __bf16* Wp  = (stg < NSTAGE_H) ? Whh : Wih;
    const int     ldw = (stg < NSTAGE_H) ? HH : VV;
    const int     kw0 = (stg < NSTAGE_H) ? stg * STEP_KSTAGE
                                         : (stg - NSTAGE_H) * STEP_KSTAGE;
    // pull this lane's weight row one stage ahead into near caches
    __builtin_prefetch(
        (const void*)(Wp + (size_t)(colW0 + (lane & 15)) * ldw + kw0 + STEP_KSTAGE), 0, 3);
#pragma unroll
    for (int ks = 0; ks < STEP_KSTAGE / 32; ++ks) {
      v16bf af0 = wmma_load_a(At,                     STEP_LDA, ks * 32, lane);
      v16bf af1 = wmma_load_a(At + 16 * STEP_LDA,     STEP_LDA, ks * 32, lane);
#pragma unroll
      for (int nt = 0; nt < 2; ++nt) {
        v16bf bf = wmma_load_b(Wp, colW0 + nt * 16, ldw, kw0 + ks * 32, lane);
        acc[0][nt] = wmma_bf16(af0, bf, acc[0][nt]);
        acc[1][nt] = wmma_bf16(af1, bf, acc[1][nt]);
      }
    }
  }

  // -------- Epilogue: accumulators -> LDS gate tile (+bias) --------
  // v8f C/D layout: VGPR r -> (M = r + 8*(lane>=16), N = lane%16)
#pragma unroll
  for (int mt = 0; mt < 2; ++mt)
#pragma unroll
    for (int nt = 0; nt < 2; ++nt)
#pragma unroll
      for (int r = 0; r < 8; ++r) {
        const int row = mt * 16 + r + (lane >> 4) * 8;
        const int col = s * 32 + nt * 16 + (lane & 15);
        gates[g][row][col] = acc[mt][nt][r] + bias[g * HH + hid0 + col];
      }
  __syncthreads();

  // -------- Fused activations + cell update (block-exclusive c slice) ------
  for (int idx = tid; idx < 32 * 64; idx += 256) {
    const int row = idx >> 6;
    const int col = idx & 63;
    const float iv = sigm(gates[0][row][col]);
    const float fv = sigm(gates[1][row][col]);
    const float gv = tanh_f(gates[2][row][col]);
    const float ov = sigm(gates[3][row][col]);
    const size_t off = (size_t)(m0 + row) * HH + hid0 + col;
    const float cn = fv * c[off] + iv * gv;
    c[off]     = cn;
    h_out[off] = (__bf16)(ov * tanh_f(cn));
  }
}

// ---------------------------------------------------------------------------
// Batched output projection: y[b,t,:] = h_all[t,b,:] @ Wout^T + b_out
// A (all 128 batch rows) async-staged through LDS, double-buffered, K=64/stage,
// same one-barrier pipeline as lstm_step.
//   grid  = (T, V/64) = (512, 4); block = 256 threads = 8 waves
//   wave w: n-tile (w&3) of the 64-col slab, batch half (w>>2), 4 m-tiles
// ---------------------------------------------------------------------------
#define OG_KSTAGE 64
#define OG_LDA    (OG_KSTAGE + 8)   // 72

__global__ __launch_bounds__(256) void out_gemm(
    const __bf16* __restrict__ h_all,  // [T, B, H] bf16
    const __bf16* __restrict__ Wout,   // [V, H]    bf16
    const float*  __restrict__ b_out,  // [V]
    float*        __restrict__ y)      // [B, T, V] fp32
{
  __shared__ __bf16 tile[2][128 * OG_LDA];     // 2 x 18 KB staged A chunks

  const int tid  = threadIdx.x;
  const int t    = blockIdx.x;
  const int nb   = blockIdx.y;
  const int wave = tid >> 5;
  const int lane = tid & 31;
  const int nt   = wave & 3;
  const int mh   = wave >> 2;
  const int col0 = nb * 64 + nt * 16;
  const __bf16* A = h_all + (size_t)t * BB * HH;

  // Staging map: thread -> (row sm, 32-element segment sk); 4 x b128 each.
  const int sm = tid >> 1;                     // 0..127
  const int sk = (tid & 1) * 32;               // 0 or 32
  const __bf16* aRow = A + (size_t)sm * HH + sk;
  const uint32_t ldsA[2] = { lds_off(&tile[0][sm * OG_LDA + sk]),
                             lds_off(&tile[1][sm * OG_LDA + sk]) };

  v8f acc[4] = {};
  const int NST = HH / OG_KSTAGE;              // 32

  auto issue = [&](int stg) {
    const __bf16* src = aRow + stg * OG_KSTAGE;
    const uint32_t l = ldsA[stg & 1];
    async_b128(l,      src);
    async_b128(l + 16, src + 8);
    async_b128(l + 32, src + 16);
    async_b128(l + 48, src + 24);
  };

  issue(0);
  for (int stg = 0; stg < NST; ++stg) {
    wait_async<0>();
    __syncthreads();
    if (stg + 1 < NST) issue(stg + 1);

    const __bf16* At = tile[stg & 1] + (mh * 64) * OG_LDA;
    const int kw0 = stg * OG_KSTAGE;
    __builtin_prefetch(
        (const void*)(Wout + (size_t)(col0 + (lane & 15)) * HH + kw0 + OG_KSTAGE), 0, 3);
#pragma unroll
    for (int ks = 0; ks < OG_KSTAGE / 32; ++ks) {
      v16bf bf = wmma_load_b(Wout, col0, HH, kw0 + ks * 32, lane);
#pragma unroll
      for (int mt = 0; mt < 4; ++mt) {
        v16bf af = wmma_load_a(At + mt * 16 * OG_LDA, OG_LDA, ks * 32, lane);
        acc[mt] = wmma_bf16(af, bf, acc[mt]);
      }
    }
  }

#pragma unroll
  for (int mt = 0; mt < 4; ++mt)
#pragma unroll
    for (int r = 0; r < 8; ++r) {
      const int m = mh * 64 + mt * 16 + r + (lane >> 4) * 8;
      const int n = col0 + (lane & 15);
      y[(size_t)m * TT * VV + (size_t)t * VV + n] = acc[mt][r] + b_out[n];
    }
}

// ---------------------------------------------------------------------------
// Launch
// ---------------------------------------------------------------------------
extern "C" void kernel_launch(void* const* d_in, const int* in_sizes, int n_in,
                              void* d_out, int out_size, void* d_ws, size_t ws_size,
                              hipStream_t stream) {
  (void)in_sizes; (void)n_in; (void)out_size; (void)ws_size;
  const float* x    = (const float*)d_in[0];
  const float* Wih  = (const float*)d_in[1];
  const float* Whh  = (const float*)d_in[2];
  const float* bih  = (const float*)d_in[3];
  const float* bhh  = (const float*)d_in[4];
  const float* Wout = (const float*)d_in[5];
  const float* bout = (const float*)d_in[6];
  float* y = (float*)d_out;

  // Workspace layout (~327 MB total)
  char* ws = (char*)d_ws;
  __bf16* Wih_b  = (__bf16*)ws;  ws += (size_t)G4 * VV * 2;       //   4 MB
  __bf16* Whh_b  = (__bf16*)ws;  ws += (size_t)G4 * HH * 2;       //  32 MB
  __bf16* Wout_b = (__bf16*)ws;  ws += (size_t)VV * HH * 2;       //   1 MB
  __bf16* x_b    = (__bf16*)ws;  ws += (size_t)BB * TT * VV * 2;  //  32 MB
  float*  bias   = (float*) ws;  ws += (size_t)G4 * 4;            //  32 KB
  __bf16* h0     = (__bf16*)ws;  ws += (size_t)BB * HH * 2;       // 512 KB
  float*  c      = (float*) ws;  ws += (size_t)BB * HH * 4;       //   1 MB
  __bf16* h_all  = (__bf16*)ws;  ws += (size_t)TT * BB * HH * 2;  // 256 MB

  const int TPB = 256;
  k_f32_to_bf16<<<(G4 * VV + TPB - 1) / TPB,      TPB, 0, stream>>>(Wih,  Wih_b,  G4 * VV);
  k_f32_to_bf16<<<(G4 * HH + TPB - 1) / TPB,      TPB, 0, stream>>>(Whh,  Whh_b,  G4 * HH);
  k_f32_to_bf16<<<(VV * HH + TPB - 1) / TPB,      TPB, 0, stream>>>(Wout, Wout_b, VV * HH);
  k_f32_to_bf16<<<(BB * TT * VV + TPB - 1) / TPB, TPB, 0, stream>>>(x,    x_b,    BB * TT * VV);
  k_bias_sum   <<<(G4 + TPB - 1) / TPB,           TPB, 0, stream>>>(bih, bhh, bias, G4);
  k_zero_state <<<(BB * HH + TPB - 1) / TPB,      TPB, 0, stream>>>(h0, c, BB * HH);

  const dim3 sgrid(HH / 64, BB / 32);  // (32, 4)
  for (int t = 0; t < TT; ++t) {
    const __bf16* hin = (t == 0) ? h0 : (h_all + (size_t)(t - 1) * BB * HH);
    __bf16* hout = h_all + (size_t)t * BB * HH;
    lstm_step<<<sgrid, 256, 0, stream>>>(x_b, Wih_b, Whh_b, bias, hin, c, hout, t);
  }

  out_gemm<<<dim3(TT, VV / 64), 256, 0, stream>>>(h_all, Wout_b, bout, y);
}
